// RealSpaceEGNNEncoder_7653631721503
// MI455X (gfx1250) — compile-verified, matrix operand order checked by reading, not run on hardware
//
#include <hip/hip_runtime.h>

typedef __attribute__((ext_vector_type(16))) _Float16 v16h;
typedef __attribute__((ext_vector_type(2)))  _Float16 h2;
typedef __attribute__((ext_vector_type(2)))  __fp16   f16x2;
typedef __attribute__((ext_vector_type(8)))  float    v8f;
typedef _Float16 half_t;

#define NNODE 32768
#define NEDGE 1048576
#define DIN   64
#define NSC   64
#define NVC   32
#define LAT   256

// ---- WMMA lane mappings (CDNA5 16x16x32 f16, wave32) ----
__device__ __host__ __forceinline__ int kmap(int t, int lane) {
  return ((lane >> 4) << 3) + ((t >> 3) << 4) + (t & 7);
}
__device__ __forceinline__ int crow(int r, int lane) { return r + ((lane >> 4) << 3); }

__device__ __forceinline__ float sigmoidf_(float x) { return 1.0f / (1.0f + __expf(-x)); }
__device__ __forceinline__ float siluf_(float x) { return x * sigmoidf_(x); }

__device__ __forceinline__ h2 pkrtz(float a, float b) {
  f16x2 r = __builtin_amdgcn_cvt_pkrtz(a, b);
  return __builtin_bit_cast(h2, r);
}

__device__ __forceinline__ v8f wmma16(v16h a, v16h b, v8f c) {
  return __builtin_amdgcn_wmma_f32_16x16x32_f16(false, a, false, b, (short)0, c, false, false);
}

union AFrag { v16h v; h2 p[8]; };

// swizzled B-fragment: one contiguous v16h per lane per fragment
__device__ __forceinline__ v16h bfrag(const half_t* __restrict__ W, int frag, int lane) {
  return ((const v16h*)W)[frag * 32 + lane];
}

// ---- weight swizzle: src f32 [L][K][N] row-major -> dst f16 fragment layout ----
__global__ void k_swz(const float* __restrict__ src, half_t* __restrict__ dst,
                      int K, int N, int L) {
  long idx = (long)blockIdx.x * blockDim.x + threadIdx.x;
  long per = (long)K * N;
  if (idx >= per * L) return;
  int l = (int)(idx / per);
  int r = (int)(idx - (long)l * per);
  int t = r & 15;
  int lane = (r >> 4) & 31;
  int q = r >> 9;
  int ntiles = N >> 4;
  int nt = q % ntiles;
  int kc = q / ntiles;
  int k = kc * 32 + kmap(t, lane);
  int n = nt * 16 + (lane & 15);
  dst[idx] = (half_t)src[(long)l * per + (long)k * N + n];
}

__global__ void k_zero(float* __restrict__ p, int n) {
  int i = blockIdx.x * blockDim.x + threadIdx.x;
  if (i < n) p[i] = 0.0f;
}

// ---- ns = x @ W_in / sqrt(64) ----
__global__ __launch_bounds__(32) void k_ns_init(const float* __restrict__ x,
    const half_t* __restrict__ Wh, float* __restrict__ ns) {
  int lane = threadIdx.x;
  int tile = blockIdx.x;
  int row = lane & 15;
  v8f acc[4] = {};
#pragma unroll
  for (int kc = 0; kc < 2; ++kc) {
    AFrag a;
#pragma unroll
    for (int h = 0; h < 8; ++h)
      a.p[h] = pkrtz(
          x[(tile * 16 + row) * DIN + kc * 32 + kmap(2 * h, lane)] * 0.125f,
          x[(tile * 16 + row) * DIN + kc * 32 + kmap(2 * h + 1, lane)] * 0.125f);
#pragma unroll
    for (int nt = 0; nt < 4; ++nt)
      acc[nt] = wmma16(a.v, bfrag(Wh, kc * 4 + nt, lane), acc[nt]);
  }
#pragma unroll
  for (int nt = 0; nt < 4; ++nt)
#pragma unroll
    for (int r = 0; r < 8; ++r)
      ns[(tile * 16 + crow(r, lane)) * NSC + nt * 16 + (lane & 15)] = acc[nt][r];
}

// ---- fused edge-message + gate + scatter : one wave per 16 edges ----
__global__ __launch_bounds__(32) void k_edge(
    const float* __restrict__ ns, const float* __restrict__ nv,
    const float* __restrict__ ea, const int* __restrict__ ei,
    const half_t* __restrict__ Wss, const half_t* __restrict__ Wvv0,
    const half_t* __restrict__ Wsv, const half_t* __restrict__ Wvs,
    const float* __restrict__ gmw, const float* __restrict__ gmb,
    float* __restrict__ a_s, float* __restrict__ a_v) {
  __shared__ float hs[16][64];
  __shared__ float hv[16][96];
  __shared__ float hvd[16][32];
  __shared__ float mv[16][96];
  __shared__ int   rid[16];
  __shared__ int   cid[16];
  __shared__ float ese[16];
  __shared__ float eve[16][3];
  __shared__ float emean[16];
  int lane = threadIdx.x;
  long tile = blockIdx.x;
  if (lane < 16) {
    long e = tile * 16 + lane;
    rid[lane] = ei[e];
    cid[lane] = ei[NEDGE + e];
    ese[lane] = ea[e * 4 + 3];
    eve[lane][0] = ea[e * 4 + 0];
    eve[lane][1] = ea[e * 4 + 1];
    eve[lane][2] = ea[e * 4 + 2];
  }
  __syncthreads();
  for (int idx = lane; idx < 16 * 64; idx += 32) {
    int e = idx >> 6, k = idx & 63;
    hs[e][k] = ns[(long)cid[e] * NSC + k];
  }
  for (int idx = lane; idx < 16 * 96; idx += 32) {
    int e = idx / 96, k = idx - e * 96;
    hv[e][k] = nv[(long)cid[e] * 96 + k];
  }
  __syncthreads();
  for (int idx = lane; idx < 16 * 32; idx += 32) {
    int e = idx >> 5, m = idx & 31;
    hvd[e][m] = hv[e][3 * m + 0] * eve[e][0] + hv[e][3 * m + 1] * eve[e][1] +
                hv[e][3 * m + 2] * eve[e][2];
  }
  __syncthreads();
  int row = lane & 15;
  const float c_ss = 0.08838834764831845f;
  const float c_vv = 0.07216878364870323f;
  const float c_sv = 0.08838834764831845f;
  const float c_vs = 0.125f;
  // pre-pack hs fragments (shared by Wss and Wsv GEMMs) into f16 registers
  h2 hs_p[2][8];
#pragma unroll
  for (int h = 0; h < 8; ++h) {
    hs_p[0][h] = pkrtz(hs[row][kmap(2 * h, lane)], hs[row][kmap(2 * h + 1, lane)]);
    hs_p[1][h] = pkrtz(hs[row][32 + kmap(2 * h, lane)], hs[row][32 + kmap(2 * h + 1, lane)]);
  }
  float esr = ese[row];
  // m_s = (hs*es)@Wss*c_ss + (hv.ev)@Wvv0*c_vv
  v8f accs[4] = {};
  {
    half_t sc = (half_t)(esr * c_ss);
#pragma unroll
    for (int kc = 0; kc < 2; ++kc) {
      AFrag a;
#pragma unroll
      for (int h = 0; h < 8; ++h) a.p[h] = sc * hs_p[kc][h];
#pragma unroll
      for (int nt = 0; nt < 4; ++nt)
        accs[nt] = wmma16(a.v, bfrag(Wss, kc * 4 + nt, lane), accs[nt]);
    }
  }
  {
    AFrag a;
#pragma unroll
    for (int h = 0; h < 8; ++h)
      a.p[h] = pkrtz(hvd[row][kmap(2 * h, lane)] * c_vv,
                     hvd[row][kmap(2 * h + 1, lane)] * c_vv);
#pragma unroll
    for (int nt = 0; nt < 4; ++nt)
      accs[nt] = wmma16(a.v, bfrag(Wvv0, nt, lane), accs[nt]);
  }
  // svec = hs@Wsv, then m_v = svec*ev + (hv_x*es)@Wvs
  v8f accv[2] = {};
  {
    half_t sc = (half_t)c_sv;
#pragma unroll
    for (int kc = 0; kc < 2; ++kc) {
      AFrag a;
#pragma unroll
      for (int h = 0; h < 8; ++h) a.p[h] = sc * hs_p[kc][h];
#pragma unroll
      for (int nt = 0; nt < 2; ++nt)
        accv[nt] = wmma16(a.v, bfrag(Wsv, kc * 2 + nt, lane), accv[nt]);
    }
  }
#pragma unroll
  for (int x = 0; x < 3; ++x) {
    v8f accm[2] = {};
    AFrag a;
    float sc = esr * c_vs;
#pragma unroll
    for (int h = 0; h < 8; ++h)
      a.p[h] = pkrtz(hv[row][3 * kmap(2 * h, lane) + x] * sc,
                     hv[row][3 * kmap(2 * h + 1, lane) + x] * sc);
#pragma unroll
    for (int nt = 0; nt < 2; ++nt)
      accm[nt] = wmma16(a.v, bfrag(Wvs, nt, lane), accm[nt]);
#pragma unroll
    for (int nt = 0; nt < 2; ++nt)
#pragma unroll
      for (int r = 0; r < 8; ++r) {
        int rr = crow(r, lane);
        int m = nt * 16 + (lane & 15);
        mv[rr][3 * m + x] = accm[nt][r] + accv[nt][r] * eve[rr][x];
      }
  }
  __syncthreads();
  if (lane < 16) {
    float s = 0.0f;
    for (int c = 0; c < 96; ++c) s += mv[lane][c];
    emean[lane] = s * (1.0f / 96.0f);
  }
  __syncthreads();
#pragma unroll
  for (int nt = 0; nt < 4; ++nt)
#pragma unroll
    for (int r = 0; r < 8; ++r) {
      int rr = crow(r, lane);
      int o = nt * 16 + (lane & 15);
      atomicAdd(&a_s[(long)rid[rr] * NSC + o], siluf_(accs[nt][r]));
    }
  for (int idx = lane; idx < 16 * 96; idx += 32) {
    int e = idx / 96, c = idx - e * 96;
    float g = sigmoidf_(emean[e] * gmw[c] + gmb[c]);
    atomicAdd(&a_v[(long)rid[e] * 96 + c], mv[e][c] * g);
  }
}

// ---- fused node bilinear update : one wave per 32 nodes (2 M-subtiles,
//      each B-fragment loaded once and used by two WMMAs) ----
__global__ __launch_bounds__(32) void k_node(
    float* __restrict__ ns, float* __restrict__ nv,
    const float* __restrict__ a_s, const float* __restrict__ a_v,
    const half_t* __restrict__ Wuss, const half_t* __restrict__ Wuvv,
    const half_t* __restrict__ Wusv, const half_t* __restrict__ Wuvs,
    const float* __restrict__ guw, const float* __restrict__ gub) {
  __shared__ float sns[32][64];
  __shared__ float sas[32][64];
  __shared__ float snv[32][96];
  __shared__ float sav[32][96];
  __shared__ float suv[32][96];
  __shared__ float smean[32];
  int lane = threadIdx.x;
  long base = (long)blockIdx.x * 32;
  for (int idx = lane; idx < 32 * 64; idx += 32) {
    int e = idx >> 6, k = idx & 63;
    sns[e][k] = ns[(base + e) * NSC + k];
    sas[e][k] = a_s[(base + e) * NSC + k];
  }
  for (int idx = lane; idx < 32 * 96; idx += 32) {
    int e = idx / 96, k = idx - e * 96;
    snv[e][k] = nv[(base + e) * 96 + k];
    sav[e][k] = a_v[(base + e) * 96 + k];
  }
  __syncthreads();
  int col = lane & 15;
  const float c_uss = 0.011048543456039806f; // 1/(64*sqrt(2))
  const float c_uvv = 0.012757756184028872f; // (1/sqrt(3))/(32*sqrt(2))
  const float c_uvx = 0.015625f;             // 1/64
  // pre-pack reused operand fragments into f16 registers (constant indexing)
  h2 as_p[2][2][8]; // [sub][j-half][h]
  h2 av_p[2][3][8]; // [sub][x][h]
#pragma unroll
  for (int s = 0; s < 2; ++s) {
    int rw = s * 16 + col;
#pragma unroll
    for (int h = 0; h < 8; ++h) {
      int k0 = kmap(2 * h, lane), k1 = kmap(2 * h + 1, lane);
      as_p[s][0][h] = pkrtz(sas[rw][k0], sas[rw][k1]);
      as_p[s][1][h] = pkrtz(sas[rw][32 + k0], sas[rw][32 + k1]);
      av_p[s][0][h] = pkrtz(sav[rw][3 * k0 + 0], sav[rw][3 * k1 + 0]);
      av_p[s][1][h] = pkrtz(sav[rw][3 * k0 + 1], sav[rw][3 * k1 + 1]);
      av_p[s][2][h] = pkrtz(sav[rw][3 * k0 + 2], sav[rw][3 * k1 + 2]);
    }
  }
  // ---- u_s ----
  v8f accs[2][4] = {};
  for (int i = 0; i < 64; ++i) { // Wuss: chunk c = 2i+half
    half_t n0 = (half_t)(sns[col][i] * c_uss);
    half_t n1 = (half_t)(sns[16 + col][i] * c_uss);
#pragma unroll
    for (int half = 0; half < 2; ++half) {
      int c = 2 * i + half;
      AFrag a0, a1;
#pragma unroll
      for (int h = 0; h < 8; ++h) {
        a0.p[h] = n0 * as_p[0][half][h];
        a1.p[h] = n1 * as_p[1][half][h];
      }
#pragma unroll
      for (int nt = 0; nt < 4; ++nt) {
        v16h b = bfrag(Wuss, c * 4 + nt, lane);
        accs[0][nt] = wmma16(a0.v, b, accs[0][nt]);
        accs[1][nt] = wmma16(a1.v, b, accs[1][nt]);
      }
    }
  }
  for (int m = 0; m < 32; ++m) { // Wuvv: chunk m, A = <nv_m, a_v_n> (packed f16 dot)
    half_t q0[2], q1[2], q2[2];
#pragma unroll
    for (int s = 0; s < 2; ++s) {
      int rw = s * 16 + col;
      q0[s] = (half_t)(snv[rw][3 * m + 0] * c_uvv);
      q1[s] = (half_t)(snv[rw][3 * m + 1] * c_uvv);
      q2[s] = (half_t)(snv[rw][3 * m + 2] * c_uvv);
    }
    AFrag a0, a1;
#pragma unroll
    for (int h = 0; h < 8; ++h) {
      a0.p[h] = q0[0] * av_p[0][0][h] + q1[0] * av_p[0][1][h] + q2[0] * av_p[0][2][h];
      a1.p[h] = q0[1] * av_p[1][0][h] + q1[1] * av_p[1][1][h] + q2[1] * av_p[1][2][h];
    }
#pragma unroll
    for (int nt = 0; nt < 4; ++nt) {
      v16h b = bfrag(Wuvv, m * 4 + nt, lane);
      accs[0][nt] = wmma16(a0.v, b, accs[0][nt]);
      accs[1][nt] = wmma16(a1.v, b, accs[1][nt]);
    }
  }
  // ---- u_v ----
#pragma unroll
  for (int x = 0; x < 3; ++x) {
    v8f accv[2][2] = {};
    for (int c = 0; c < 64; ++c) { // Wusv: k = c*32 + n
      half_t n0 = (half_t)(sns[col][c] * c_uvx);
      half_t n1 = (half_t)(sns[16 + col][c] * c_uvx);
      AFrag a0, a1;
#pragma unroll
      for (int h = 0; h < 8; ++h) {
        a0.p[h] = n0 * av_p[0][x][h];
        a1.p[h] = n1 * av_p[1][x][h];
      }
#pragma unroll
      for (int nt = 0; nt < 2; ++nt) {
        v16h b = bfrag(Wusv, c * 2 + nt, lane);
        accv[0][nt] = wmma16(a0.v, b, accv[0][nt]);
        accv[1][nt] = wmma16(a1.v, b, accv[1][nt]);
      }
    }
    for (int i = 0; i < 32; ++i) { // Wuvs: chunk c = 2i+half
      half_t v0 = (half_t)(snv[col][3 * i + x] * c_uvx);
      half_t v1 = (half_t)(snv[16 + col][3 * i + x] * c_uvx);
#pragma unroll
      for (int half = 0; half < 2; ++half) {
        int c = 2 * i + half;
        AFrag a0, a1;
#pragma unroll
        for (int h = 0; h < 8; ++h) {
          a0.p[h] = v0 * as_p[0][half][h];
          a1.p[h] = v1 * as_p[1][half][h];
        }
#pragma unroll
        for (int nt = 0; nt < 2; ++nt) {
          v16h b = bfrag(Wuvs, c * 2 + nt, lane);
          accv[0][nt] = wmma16(a0.v, b, accv[0][nt]);
          accv[1][nt] = wmma16(a1.v, b, accv[1][nt]);
        }
      }
    }
#pragma unroll
    for (int s = 0; s < 2; ++s)
#pragma unroll
      for (int nt = 0; nt < 2; ++nt)
#pragma unroll
        for (int r = 0; r < 8; ++r) {
          int rw = s * 16 + crow(r, lane);
          int o = nt * 16 + (lane & 15);
          suv[rw][3 * o + x] = accv[s][nt][r];
        }
  }
  __syncthreads();
  {
    float ssum = 0.0f;
    for (int c = 0; c < 96; ++c) ssum += suv[lane][c];
    smean[lane] = ssum * (1.0f / 96.0f);
  }
  __syncthreads();
#pragma unroll
  for (int s = 0; s < 2; ++s)
#pragma unroll
    for (int nt = 0; nt < 4; ++nt)
#pragma unroll
      for (int r = 0; r < 8; ++r) {
        int rw = s * 16 + crow(r, lane);
        int o = nt * 16 + (lane & 15);
        ns[(base + rw) * NSC + o] = sns[rw][o] + siluf_(accs[s][nt][r]);
      }
  for (int idx = lane; idx < 32 * 96; idx += 32) {
    int e = idx / 96, c = idx - e * 96;
    float g = sigmoidf_(smean[e] * guw[c] + gub[c]);
    nv[(base + e) * 96 + c] = snv[e][c] + suv[e][c] * g;
  }
}

// ---- out = ns @ W_out / sqrt(64) ----
__global__ __launch_bounds__(32) void k_out(const float* __restrict__ ns,
    const half_t* __restrict__ Wouth, float* __restrict__ out) {
  int lane = threadIdx.x;
  int tile = blockIdx.x;
  int rt = tile >> 2;
  int cgt = (tile & 3) * 4;
  int row = lane & 15;
  v8f acc[4] = {};
#pragma unroll
  for (int kc = 0; kc < 2; ++kc) {
    AFrag a;
#pragma unroll
    for (int h = 0; h < 8; ++h)
      a.p[h] = pkrtz(
          ns[(rt * 16 + row) * NSC + kc * 32 + kmap(2 * h, lane)] * 0.125f,
          ns[(rt * 16 + row) * NSC + kc * 32 + kmap(2 * h + 1, lane)] * 0.125f);
#pragma unroll
    for (int nt = 0; nt < 4; ++nt)
      acc[nt] = wmma16(a.v, bfrag(Wouth, kc * 16 + cgt + nt, lane), acc[nt]);
  }
#pragma unroll
  for (int nt = 0; nt < 4; ++nt)
#pragma unroll
    for (int r = 0; r < 8; ++r)
      out[(long)(rt * 16 + crow(r, lane)) * LAT + (cgt + nt) * 16 + (lane & 15)] = acc[nt][r];
}

extern "C" void kernel_launch(void* const* d_in, const int* in_sizes, int n_in,
                              void* d_out, int out_size, void* d_ws, size_t ws_size,
                              hipStream_t stream) {
  (void)in_sizes; (void)n_in; (void)out_size; (void)ws_size;
  const float* x    = (const float*)d_in[0];
  const float* ea   = (const float*)d_in[1];
  const float* Win  = (const float*)d_in[2];
  const float* Wout = (const float*)d_in[3];
  const float* Wss  = (const float*)d_in[4];
  const float* Wvv0 = (const float*)d_in[5];
  const float* Wsv  = (const float*)d_in[6];
  const float* Wvs  = (const float*)d_in[7];
  const float* gmw  = (const float*)d_in[8];
  const float* gmb  = (const float*)d_in[9];
  const float* Wuss = (const float*)d_in[10];
  const float* Wuvv = (const float*)d_in[11];
  const float* Wusv = (const float*)d_in[12];
  const float* Wuvs = (const float*)d_in[13];
  const float* guw  = (const float*)d_in[14];
  const float* gub  = (const float*)d_in[15];
  const int*   ei   = (const int*)d_in[16];

  char* ws = (char*)d_ws;
  size_t off = 0;
  auto alloc = [&](size_t bytes) {
    void* p = ws + off;
    off += (bytes + 255) & ~(size_t)255;
    return p;
  };
  float* ns  = (float*)alloc((size_t)NNODE * 64 * 4);
  float* nv  = (float*)alloc((size_t)NNODE * 96 * 4);
  float* a_s = (float*)alloc((size_t)NNODE * 64 * 4);
  float* a_v = (float*)alloc((size_t)NNODE * 96 * 4);
  half_t* Winh  = (half_t*)alloc(4096 * 2);
  half_t* Wssh  = (half_t*)alloc(2 * 4096 * 2);
  half_t* Wvv0h = (half_t*)alloc(2 * 2048 * 2);
  half_t* Wsvh  = (half_t*)alloc(2 * 2048 * 2);
  half_t* Wvsh  = (half_t*)alloc(2 * 1024 * 2);
  half_t* Wussh = (half_t*)alloc(2 * 262144 * 2);
  half_t* Wuvvh = (half_t*)alloc(2 * 65536 * 2);
  half_t* Wusvh = (half_t*)alloc(2 * 65536 * 2);
  half_t* Wuvsh = (half_t*)alloc(2 * 65536 * 2);
  half_t* Wouth = (half_t*)alloc(16384 * 2);

  auto swz = [&](const float* s, half_t* d, int K, int N, int L) {
    long tot = (long)K * N * L;
    k_swz<<<(int)((tot + 255) / 256), 256, 0, stream>>>(s, d, K, N, L);
  };
  swz(Win,  Winh,  64,   64,  1);
  swz(Wss,  Wssh,  64,   64,  2);
  swz(Wvv0, Wvv0h, 32,   64,  2);
  swz(Wsv,  Wsvh,  64,   32,  2);
  swz(Wvs,  Wvsh,  32,   32,  2);
  swz(Wuss, Wussh, 4096, 64,  2);
  swz(Wuvv, Wuvvh, 1024, 64,  2);
  swz(Wusv, Wusvh, 2048, 32,  2);
  swz(Wuvs, Wuvsh, 2048, 32,  2);
  swz(Wout, Wouth, 64,   256, 1);

  k_zero<<<(NNODE * 96 + 255) / 256, 256, 0, stream>>>(nv, NNODE * 96);
  k_ns_init<<<NNODE / 16, 32, 0, stream>>>(x, Winh, ns);

  for (int l = 0; l < 2; ++l) {
    k_zero<<<(NNODE * 64 + 255) / 256, 256, 0, stream>>>(a_s, NNODE * 64);
    k_zero<<<(NNODE * 96 + 255) / 256, 256, 0, stream>>>(a_v, NNODE * 96);
    k_edge<<<NEDGE / 16, 32, 0, stream>>>(ns, nv, ea, ei,
        Wssh + l * 4096, Wvv0h + l * 2048, Wsvh + l * 2048, Wvsh + l * 1024,
        gmw + l * 96, gmb + l * 96, a_s, a_v);
    k_node<<<NNODE / 32, 32, 0, stream>>>(ns, nv, a_s, a_v,
        Wussh + l * 262144, Wuvvh + l * 65536, Wusvh + l * 65536, Wuvsh + l * 65536,
        guw + l * 96, gub + l * 96);
  }
  k_out<<<(NNODE / 16) * 4, 32, 0, stream>>>(ns, Wouth, (float*)d_out);
}
